// BinarySquareClassifier_91139206021420
// MI455X (gfx1250) — matrix operand order repaired and checked by reading, not compiled, not run on hardware
//
#include <hip/hip_runtime.h>
#include <hip/hip_bf16.h>

// ---------------------------------------------------------------------------
// SNN (3-layer LIF, T=1024 sequential) for MI455X / gfx1250, wave32 + WMMA.
//
// Kernel 1 (repack_x): x fp32 [B=2048, C=30, T=1024] -> fp16, pre-swizzled
//   into the 16-bit WMMA A-matrix lane/VGPR layout (ISA 7.12.2), C padded
//   to 32 with zeros. Fully parallel, coalesced both ways via LDS staging.
//
// Kernel 2 (snn_main): 128 blocks x 32 threads (1 wave = 16 batch rows).
//   Per timestep: 9x v_wmma_f32_16x16x32_f16; LIF fused into the WMMA C
//   operand (C_in = 0.9*m - reset + bias  =>  m_new = A*B + C_in).
//   Spike C-layout -> A-layout conversion through per-wave LDS with
//   16B-aligned ds_load_b128 reads. All membranes live in VGPRs.
// ---------------------------------------------------------------------------

typedef __attribute__((ext_vector_type(16))) _Float16 v16h;
typedef __attribute__((ext_vector_type(8)))  _Float16 h8_t;
typedef __attribute__((ext_vector_type(8)))  float    v8f;

#define SNN_B     2048
#define SNN_C     30
#define SNN_T     1024
#define SNN_NW    (SNN_B / 16)   // 128 waves / 16-row tiles
#define TCHUNK    32

// ---------------------------------------------------------------------------
// Repack x into A-matrix order. Output element for (w, t, lane L, half j):
//   row = L & 15
//   K   = 16*(j>>3) + 8*(L>>4) + (j&7)      (16-bit A layout, 16x32 tile)
//   val = (K < 30) ? (half)x[16w+row, K, t] : 0
// Stored at xp[(((w*1024 + t)*32 + L)*16 + j]  -> 32B per lane, coalesced.
// ---------------------------------------------------------------------------
__global__ __launch_bounds__(256) void repack_x(const float* __restrict__ x,
                                                _Float16* __restrict__ xp) {
  const int w  = blockIdx.x;            // 0..127 (16-row tile)
  const int t0 = blockIdx.y * TCHUNK;   // time chunk base
  const int tid = threadIdx.x;

  __shared__ __align__(16) _Float16 tile[16 * 32 * TCHUNK]; // [row][c:32][t]

  // zero-pad channels 30,31
  for (int f = tid; f < 16 * 2 * TCHUNK; f += 256) {
    const int tt = f % TCHUNK;
    const int rc = f / TCHUNK;
    const int row = rc >> 1;
    const int c   = 30 + (rc & 1);
    tile[(row * 32 + c) * TCHUNK + tt] = (_Float16)0.0f;
  }
  // stage real channels, coalesced along t
  for (int f = tid; f < 16 * 30 * TCHUNK; f += 256) {
    const int tt = f % TCHUNK;
    const int rc = f / TCHUNK;
    const int row = rc / 30;
    const int c   = rc % 30;
    const float v = x[((size_t)(w * 16 + row) * SNN_C + c) * SNN_T + t0 + tt];
    tile[(row * 32 + c) * TCHUNK + tt] = (_Float16)v;
  }
  __syncthreads();

  // emit packed A-operand bytes, coalesced along lane L
  for (int f = tid; f < TCHUNK * 32; f += 256) {
    const int L  = f & 31;
    const int tt = f >> 5;
    const int row = L & 15;
    const int kA  = (L >> 4) * 8;
    union { _Float16 h[16]; uint4 q[2]; } u;
#pragma unroll
    for (int j = 0; j < 16; ++j) {
      const int k = ((j >> 3) << 4) + kA + (j & 7);
      u.h[j] = tile[(row * 32 + k) * TCHUNK + tt];
    }
    uint4* dst = (uint4*)(xp + ((size_t)(w * SNN_T + t0 + tt) * 32 + L) * 16);
    dst[0] = u.q[0];
    dst[1] = u.q[1];
  }
}

// ---------------------------------------------------------------------------
// Persistent sequential SNN kernel: one wave per 16 batch rows.
// ---------------------------------------------------------------------------
__global__ __launch_bounds__(32) void snn_main(const _Float16* __restrict__ xp,
                                               const float* __restrict__ W1,
                                               const float* __restrict__ b1,
                                               const float* __restrict__ W2,
                                               const float* __restrict__ b2,
                                               const float* __restrict__ W3,
                                               const float* __restrict__ b3,
                                               float* __restrict__ out) {
  const int w   = blockIdx.x;       // 0..127
  const int L   = threadIdx.x;      // 0..31
  const int n16 = L & 15;           // B-matrix column / C-matrix N
  const int kA  = (L >> 4) * 8;     // A-operand K sub-base (per 16-K group)
  const int kB  = (L >> 4) * 8;     // B-operand K sub-base (per 16-K group, matches A)
  const int rowOff = (L >> 4) * 8;  // C-layout: lanes 16-31 hold M = v + 8

  // ---- one-time weight gather into B-matrix register layout -------------
  v16h w1b[4];                       // layer 1: B = W1^T (30x64), 4 N-tiles
#pragma unroll
  for (int tc = 0; tc < 4; ++tc) {
    const int neuron = tc * 16 + n16;
#pragma unroll
    for (int j = 0; j < 16; ++j) {
      const int k = kB + j + ((j >> 3) << 3);
      w1b[tc][j] = (k < SNN_C) ? (_Float16)W1[neuron * SNN_C + k] : (_Float16)0.0f;
    }
  }
  v16h w2b[2][2];                    // layer 2: B = W2^T (64x32), [Ntile][Ktile]
#pragma unroll
  for (int nt = 0; nt < 2; ++nt) {
    const int neuron = nt * 16 + n16;
#pragma unroll
    for (int kt = 0; kt < 2; ++kt) {
#pragma unroll
      for (int j = 0; j < 16; ++j)
        w2b[nt][kt][j] = (_Float16)W2[neuron * 64 + kt * 32 + kB + j + ((j >> 3) << 3)];
    }
  }
  v16h w3b;                          // layer 3: B = W3^T (32x2), N padded to 16
#pragma unroll
  for (int j = 0; j < 16; ++j)
    w3b[j] = (n16 < 2) ? (_Float16)W3[n16 * 32 + kB + j + ((j >> 3) << 3)] : (_Float16)0.0f;

  float b1v[4], b2v[2];
#pragma unroll
  for (int i = 0; i < 4; ++i) b1v[i] = b1[i * 16 + n16];
#pragma unroll
  for (int i = 0; i < 2; ++i) b2v[i] = b2[i * 16 + n16];
  const float b3v = (n16 < 2) ? b3[n16] : 0.0f;

  // ---- state (membranes + accumulator) entirely in VGPRs ----------------
  v8f m1[4] = {{}, {}, {}, {}};
  v8f m2[2] = {{}, {}};
  v8f m3 = {};
  v8f acc = {};

  // per-wave LDS bounce buffers for C-layout -> A-layout spike transpose
  __shared__ __align__(16) _Float16 s1buf[16 * 64];
  __shared__ __align__(16) _Float16 s2buf[16 * 32];

  const _Float16* xw = xp + ((size_t)w * SNN_T * 32 + L) * 16;
  const int arow = n16;              // A-layout: this lane's matrix row

  for (int t = 0; t < SNN_T; ++t) {
    if (t + 8 < SNN_T)
      __builtin_prefetch(xw + (size_t)(t + 8) * 512, 0, 3); // global_prefetch_b8

    // ---- layer 1: m1 = x_t @ W1^T + (0.9 m1 - reset + b1) --------------
    const v16h a1 = *(const v16h*)(xw + (size_t)t * 512);
#pragma unroll
    for (int tc = 0; tc < 4; ++tc) {
      v8f c;
#pragma unroll
      for (int i = 0; i < 8; ++i) {
        const float m = m1[tc][i];
        const float r = (m > 1.0f) ? 1.0f : 0.0f;   // reset from previous mem
        c[i] = 0.9f * m - r + b1v[tc];
      }
      m1[tc] = __builtin_amdgcn_wmma_f32_16x16x32_f16(
          false, a1, false, w1b[tc], (short)0, c, false, false);
#pragma unroll
      for (int i = 0; i < 8; ++i) {
        const float s = (m1[tc][i] > 1.0f) ? 1.0f : 0.0f;
        s1buf[(i + rowOff) * 64 + tc * 16 + n16] = (_Float16)s;
      }
    }
    __builtin_amdgcn_wave_barrier();   // DS in-order within wave; pin schedule

    // gather s1 as A-matrix operands (two aligned 16B LDS reads per k-tile)
    v16h a2[2];
#pragma unroll
    for (int kt = 0; kt < 2; ++kt) {
      union { v16h v; h8_t h[2]; } u;
      u.h[0] = *(const h8_t*)&s1buf[arow * 64 + kt * 32 + kA];
      u.h[1] = *(const h8_t*)&s1buf[arow * 64 + kt * 32 + 16 + kA];
      a2[kt] = u.v;
    }

    // ---- layer 2: K=64 -> two chained WMMAs per N-tile ------------------
#pragma unroll
    for (int nt = 0; nt < 2; ++nt) {
      v8f c;
#pragma unroll
      for (int i = 0; i < 8; ++i) {
        const float m = m2[nt][i];
        const float r = (m > 1.0f) ? 1.0f : 0.0f;
        c[i] = 0.9f * m - r + b2v[nt];
      }
      c = __builtin_amdgcn_wmma_f32_16x16x32_f16(
          false, a2[0], false, w2b[nt][0], (short)0, c, false, false);
      m2[nt] = __builtin_amdgcn_wmma_f32_16x16x32_f16(
          false, a2[1], false, w2b[nt][1], (short)0, c, false, false);
#pragma unroll
      for (int i = 0; i < 8; ++i) {
        const float s = (m2[nt][i] > 1.0f) ? 1.0f : 0.0f;
        s2buf[(i + rowOff) * 32 + nt * 16 + n16] = (_Float16)s;
      }
    }
    __builtin_amdgcn_wave_barrier();

    // ---- layer 3: K=32, single WMMA (N padded 2->16 with zero weights) --
    union { v16h v; h8_t h[2]; } u3;
    u3.h[0] = *(const h8_t*)&s2buf[arow * 32 + kA];
    u3.h[1] = *(const h8_t*)&s2buf[arow * 32 + 16 + kA];
    v8f c3;
#pragma unroll
    for (int i = 0; i < 8; ++i) {
      const float m = m3[i];
      const float r = (m > 1.0f) ? 1.0f : 0.0f;
      c3[i] = 0.9f * m - r + b3v;
    }
    m3 = __builtin_amdgcn_wmma_f32_16x16x32_f16(
        false, u3.v, false, w3b, (short)0, c3, false, false);
#pragma unroll
    for (int i = 0; i < 8; ++i)
      acc[i] += (m3[i] > 1.0f) ? 1.0f : 0.0f;
  }

  // ---- write accumulated output spikes [B, 2] ---------------------------
  if (n16 < 2) {
#pragma unroll
    for (int i = 0; i < 8; ++i) {
      const int row = w * 16 + rowOff + i;
      out[row * 2 + n16] = acc[i];
    }
  }
}

// ---------------------------------------------------------------------------
extern "C" void kernel_launch(void* const* d_in, const int* in_sizes, int n_in,
                              void* d_out, int out_size, void* d_ws, size_t ws_size,
                              hipStream_t stream) {
  const float* x  = (const float*)d_in[0];
  const float* W1 = (const float*)d_in[1];
  const float* b1 = (const float*)d_in[2];
  const float* W2 = (const float*)d_in[3];
  const float* b2 = (const float*)d_in[4];
  const float* W3 = (const float*)d_in[5];
  const float* b3 = (const float*)d_in[6];
  float* out = (float*)d_out;

  // workspace: packed fp16 A-operands, 128 * 1024 * 32 * 16 halves = 128 MiB
  _Float16* xp = (_Float16*)d_ws;

  dim3 g1(SNN_NW, SNN_T / TCHUNK);
  repack_x<<<g1, 256, 0, stream>>>(x, xp);
  snn_main<<<SNN_NW, 32, 0, stream>>>(xp, W1, b1, W2, b2, W3, b3, out);
}
